// Decoder_45414984188138
// MI455X (gfx1250) — matrix operand order repaired and checked by reading, not compiled, not run on hardware
//
#include <hip/hip_runtime.h>
#include <hip/hip_bf16.h>
#include <stdint.h>

#define D_MODEL 1024
#define N_HEADS 16
#define DKHEAD  64
#define D_FF    4096
#define BATCH   2
#define SEQ     2048
#define NTOK    (BATCH * SEQ)   // 4096

typedef __attribute__((ext_vector_type(16))) __bf16 bf16x16;
typedef __attribute__((ext_vector_type(8)))  float  f32x8;

// ---------------------------------------------------------------- helpers ---

__device__ __forceinline__ unsigned short f2bf(float x) {
  unsigned int u = __float_as_uint(x);
  unsigned int r = u + 0x7FFFu + ((u >> 16) & 1u);   // round-to-nearest-even
  return (unsigned short)(r >> 16);
}

__device__ __forceinline__ f32x8 wmma_bf16(bf16x16 a, bf16x16 b, f32x8 c) {
  // (neg_a, A, neg_b, B, c_mod, C, reuse_a, reuse_b)
  return __builtin_amdgcn_wmma_f32_16x16x32_bf16(false, a, false, b, (short)0, c,
                                                 false, false);
}

// Low 32 bits of a flat pointer into LDS == group-segment byte offset
// (flat LDS aperture: the address's low 32 bits).
__device__ __forceinline__ unsigned lds_off_of(const void* p) {
  return (unsigned)(unsigned long long)(uintptr_t)p;
}

// GLOBAL_LOAD_ASYNC_TO_LDS_B128: 16B/lane memory -> LDS, tracked by ASYNCcnt.
__device__ __forceinline__ void async_load_b128(unsigned lds_off, const void* gaddr) {
  asm volatile("global_load_async_to_lds_b128 %0, %1, off"
               :: "v"(lds_off), "v"((unsigned long long)(uintptr_t)gaddr)
               : "memory");
}

__device__ __forceinline__ void wait_async0() {
  asm volatile("s_wait_asynccnt 0" ::: "memory");
}

// A-fragment (16 rows x K=32 bf16) from row-major storage, row stride `stride`
// elements. ISA layout: lane L holds row M=L%16; K chunks {hi*8..+7, 16+hi*8..+7}.
__device__ __forceinline__ bf16x16 load_fragA(const unsigned short* base, int row0,
                                              int stride, int k0) {
  int lane = threadIdx.x & 31;
  int m = lane & 15, hi = lane >> 4;
  const unsigned short* p = base + (size_t)(row0 + m) * stride + k0 + hi * 8;
  union { bf16x16 v; uint4 q[2]; } f;
  f.q[0] = *(const uint4*)(p);
  f.q[1] = *(const uint4*)(p + 16);
  return f.v;
}

// B-fragment (K=32 x 16 cols) sourced from B^T stored row-major [N][K].
// ISA layout: lane L holds col N=L%16; K chunks {hi*8..+7, 16+hi*8..+7}.
__device__ __forceinline__ bf16x16 load_fragB(const unsigned short* base, int n0,
                                              int stride, int k0) {
  int lane = threadIdx.x & 31;
  int n = lane & 15, hi = lane >> 4;
  const unsigned short* p = base + (size_t)(n0 + n) * stride + k0 + hi * 8;
  union { bf16x16 v; uint4 q[2]; } f;
  f.q[0] = *(const uint4*)(p);
  f.q[1] = *(const uint4*)(p + 16);
  return f.v;
}

// -------------------------------------------------------- repack / convert ---

// [H, D, d_k] f32 -> packed B^T bf16 [N=H*d_k][K=D]
__global__ void pack_heads(const float* __restrict__ W, unsigned short* __restrict__ Wt) {
  int idx = blockIdx.x * 256 + threadIdx.x;            // idx = n*1024 + d
  int n = idx >> 10, d = idx & 1023;
  int h = n >> 6, k = n & 63;
  Wt[idx] = f2bf(W[(size_t)h * D_MODEL * DKHEAD + (size_t)d * DKHEAD + k]);
}

// W [R][C] f32 -> Wt [C][R] bf16
__global__ void transpose_cvt(const float* __restrict__ W, unsigned short* __restrict__ Wt,
                              int R, int C) {
  unsigned idx = blockIdx.x * 256 + threadIdx.x;       // output index: c*R + r
  unsigned c = idx / (unsigned)R, r = idx % (unsigned)R;
  Wt[idx] = f2bf(W[(size_t)r * C + c]);
}

__global__ void cvt_bf16(const float* __restrict__ x, unsigned short* __restrict__ y, int n) {
  int idx = blockIdx.x * 256 + threadIdx.x;
  if (idx < n) y[idx] = f2bf(x[idx]);
}

// V [B*Tk][D] (head-interleaved) bf16 -> Vt [b*H+h][64][Tk] bf16 (d_k-major)
__global__ void transpose_v_heads(const unsigned short* __restrict__ V,
                                  unsigned short* __restrict__ Vt) {
  unsigned idx = blockIdx.x * 256 + threadIdx.x;       // (bh*64 + k)*SEQ + s
  unsigned s  = idx & (SEQ - 1);
  unsigned k  = (idx >> 11) & 63;
  unsigned bh = idx >> 17;
  unsigned b = bh >> 4, h = bh & 15;
  Vt[idx] = V[((size_t)b * SEQ + s) * D_MODEL + h * 64 + k];
}

// ------------------------------------------------------------- WMMA GEMM ----
// C[M,N] = act(A[M,K] @ B[K,N] + bias),  A bf16 row-major, Bt bf16 [N][K].
// 128x128 block tile, BK=32, 8 waves: each wave = 4 (M) x 2 (N) 16x16 tiles.
// Double-buffered LDS staged with GLOBAL_LOAD_ASYNC_TO_LDS_B128.

#define BM 128
#define BN 128
#define BK 32

__device__ __forceinline__ void stage_tiles_async(
    const unsigned short* __restrict__ A, const unsigned short* __restrict__ Bt,
    int K, int bm, int bn, int k0, unsigned short* As, unsigned short* Bs) {
#pragma unroll
  for (int c = 0; c < 2; ++c) {
    int chunk = threadIdx.x * 2 + c;           // 0..511 (256 threads)
    int r = chunk >> 2, cc = (chunk & 3) * 8;  // 128 rows x 4 chunks of 8 bf16
    async_load_b128(lds_off_of(&As[r * BK + cc]), &A[(size_t)(bm + r) * K + k0 + cc]);
    async_load_b128(lds_off_of(&Bs[r * BK + cc]), &Bt[(size_t)(bn + r) * K + k0 + cc]);
  }
}

template <int RELU, bool WF, bool WB>
__global__ __launch_bounds__(256) void gemm_bf16(
    const unsigned short* __restrict__ A, const unsigned short* __restrict__ Bt,
    const float* __restrict__ bias, float* __restrict__ Cf,
    unsigned short* __restrict__ Cb, int M, int N, int K) {
  __shared__ unsigned short As[2][BM * BK];
  __shared__ unsigned short Bs[2][BN * BK];

  const int bm = blockIdx.y * BM;
  const int bn = blockIdx.x * BN;
  const int wave = threadIdx.x >> 5;
  const int wm = wave >> 2;          // 0..1 : 64-row slab
  const int wn = wave & 3;           // 0..3 : 32-col slab
  const int lane = threadIdx.x & 31;
  const int n16 = lane & 15, hi = lane >> 4;

  f32x8 acc[4][2];
#pragma unroll
  for (int i = 0; i < 4; ++i)
#pragma unroll
    for (int j = 0; j < 2; ++j)
#pragma unroll
      for (int v = 0; v < 8; ++v) acc[i][j][v] = 0.f;

  stage_tiles_async(A, Bt, K, bm, bn, 0, As[0], Bs[0]);
  wait_async0();
  __syncthreads();

  int p = 0;
  for (int k0 = 0; k0 < K; k0 += BK) {
    const int kn = k0 + BK;
    if (kn < K) stage_tiles_async(A, Bt, K, bm, bn, kn, As[p ^ 1], Bs[p ^ 1]);

    bf16x16 af[4], bfr[2];
#pragma unroll
    for (int i = 0; i < 4; ++i) af[i] = load_fragA(As[p], wm * 64 + i * 16, BK, 0);
#pragma unroll
    for (int j = 0; j < 2; ++j) bfr[j] = load_fragB(Bs[p], wn * 32 + j * 16, BK, 0);
#pragma unroll
    for (int i = 0; i < 4; ++i)
#pragma unroll
      for (int j = 0; j < 2; ++j) acc[i][j] = wmma_bf16(af[i], bfr[j], acc[i][j]);

    if (kn < K) wait_async0();
    __syncthreads();
    p ^= 1;
  }

#pragma unroll
  for (int i = 0; i < 4; ++i)
#pragma unroll
    for (int j = 0; j < 2; ++j) {
      const int col = bn + wn * 32 + j * 16 + n16;
      const float bv = bias[col];
#pragma unroll
      for (int v = 0; v < 8; ++v) {
        const int row = bm + wm * 64 + i * 16 + v + 8 * hi;
        float x = acc[i][j][v] + bv;
        if (RELU) x = fmaxf(x, 0.f);
        if (WF) Cf[(size_t)row * N + col] = x;
        if (WB) Cb[(size_t)row * N + col] = f2bf(x);
      }
    }
}

// -------------------------------------------------------- flash attention ---
// grid: (SEQ/64, BATCH*N_HEADS), 128 threads (4 waves, 16 query rows each).
// Q/K: bf16 [NTOK][D_MODEL] head-interleaved; Vt: bf16 [b*H+h][64][Tk].

__global__ __launch_bounds__(128) void flash_attn(
    const unsigned short* __restrict__ Q, const unsigned short* __restrict__ Km,
    const unsigned short* __restrict__ Vt, float* __restrict__ O, int Tk, int causal) {
  __shared__ unsigned short Qs[64 * 64];
  __shared__ unsigned short Ps[4][16 * 32];

  const int b = blockIdx.y >> 4;
  const int h = blockIdx.y & 15;
  const int wave = threadIdx.x >> 5;
  const int lane = threadIdx.x & 31;
  const int n16 = lane & 15, hi = lane >> 4;

  const size_t qrow_base = (size_t)b * SEQ + (size_t)blockIdx.x * 64;
  for (int c = threadIdx.x; c < 512; c += 128) {
    int r = c >> 3, cc = (c & 7) * 8;
    async_load_b128(lds_off_of(&Qs[r * 64 + cc]),
                    &Q[(qrow_base + r) * D_MODEL + h * 64 + cc]);
  }
  wait_async0();
  __syncthreads();

  const int q0 = blockIdx.x * 64 + wave * 16;
  bf16x16 aq0 = load_fragA(Qs, wave * 16, 64, 0);
  bf16x16 aq1 = load_fragA(Qs, wave * 16, 64, 32);

  const unsigned short* vtb = Vt + (size_t)(b * N_HEADS + h) * 64 * Tk;

  f32x8 o[4];
  float m[8], l[8];
#pragma unroll
  for (int nt = 0; nt < 4; ++nt)
#pragma unroll
    for (int v = 0; v < 8; ++v) o[nt][v] = 0.f;
#pragma unroll
  for (int v = 0; v < 8; ++v) { m[v] = -3.0e38f; l[v] = 0.f; }

  const int nb = causal ? ((q0 + 15) / 32 + 1) : (Tk / 32);
  for (int j = 0; j < nb; ++j) {
    const int s0 = j * 32;
    const unsigned short* kbase = Km + ((size_t)b * Tk + s0) * D_MODEL + h * 64;
    bf16x16 bk00 = load_fragB(kbase, 0, D_MODEL, 0);
    bf16x16 bk01 = load_fragB(kbase, 0, D_MODEL, 32);
    bf16x16 bk10 = load_fragB(kbase, 16, D_MODEL, 0);
    bf16x16 bk11 = load_fragB(kbase, 16, D_MODEL, 32);

    f32x8 c0, c1;
#pragma unroll
    for (int v = 0; v < 8; ++v) { c0[v] = 0.f; c1[v] = 0.f; }
    c0 = wmma_bf16(aq0, bk00, c0);
    c0 = wmma_bf16(aq1, bk01, c0);
    c1 = wmma_bf16(aq0, bk10, c1);
    c1 = wmma_bf16(aq1, bk11, c1);

    // scale, causal mask, online softmax; C layout: row = v + 8*hi, col = n16
#pragma unroll
    for (int v = 0; v < 8; ++v) {
      float s0v = c0[v] * 0.125f;   // 1/sqrt(64)
      float s1v = c1[v] * 0.125f;
      if (causal) {
        int qrow = q0 + v + 8 * hi;
        if (s0 + n16 > qrow)      s0v = -3.0e38f;
        if (s0 + 16 + n16 > qrow) s1v = -3.0e38f;
      }
      float rm = fmaxf(s0v, s1v);
      rm = fmaxf(rm, __shfl_xor(rm, 1, 32));
      rm = fmaxf(rm, __shfl_xor(rm, 2, 32));
      rm = fmaxf(rm, __shfl_xor(rm, 4, 32));
      rm = fmaxf(rm, __shfl_xor(rm, 8, 32));
      float mn = fmaxf(m[v], rm);
      float corr = __expf(m[v] - mn);
      m[v] = mn;
      float p0 = __expf(s0v - mn);
      float p1 = __expf(s1v - mn);
      float rs = p0 + p1;
      rs += __shfl_xor(rs, 1, 32);
      rs += __shfl_xor(rs, 2, 32);
      rs += __shfl_xor(rs, 4, 32);
      rs += __shfl_xor(rs, 8, 32);
      l[v] = l[v] * corr + rs;
#pragma unroll
      for (int nt = 0; nt < 4; ++nt) o[nt][v] *= corr;
      int mrow = v + 8 * hi;
      Ps[wave][mrow * 32 + n16]      = f2bf(p0);
      Ps[wave][mrow * 32 + 16 + n16] = f2bf(p1);
    }

    bf16x16 pa = load_fragA(&Ps[wave][0], 0, 32, 0);
#pragma unroll
    for (int nt = 0; nt < 4; ++nt) {
      bf16x16 bv = load_fragB(vtb, nt * 16, Tk, s0);   // V^T rows = d_k, cols = s
      o[nt] = wmma_bf16(pa, bv, o[nt]);
    }
  }

  float invl[8];
#pragma unroll
  for (int v = 0; v < 8; ++v) invl[v] = 1.0f / l[v];
#pragma unroll
  for (int nt = 0; nt < 4; ++nt)
#pragma unroll
    for (int v = 0; v < 8; ++v) {
      size_t row = qrow_base + wave * 16 + v + 8 * hi;
      O[row * D_MODEL + h * 64 + nt * 16 + n16] = o[nt][v] * invl[v];
    }
}

// --------------------------------------------------- residual + layernorm ---

__global__ __launch_bounds__(256) void resid_ln(
    const float* __restrict__ A, const float* __restrict__ Bv,
    const float* __restrict__ gamma, const float* __restrict__ beta,
    float* __restrict__ outf, unsigned short* __restrict__ outb) {
  __shared__ float red[16];
  const size_t base = (size_t)blockIdx.x * D_MODEL;
  float x[4]; float s = 0.f, s2 = 0.f;
#pragma unroll
  for (int i = 0; i < 4; ++i) {
    int c = threadIdx.x + i * 256;
    x[i] = A[base + c] + Bv[base + c];
    s += x[i]; s2 += x[i] * x[i];
  }
#pragma unroll
  for (int off = 1; off < 32; off <<= 1) {
    s  += __shfl_xor(s,  off, 32);
    s2 += __shfl_xor(s2, off, 32);
  }
  int wave = threadIdx.x >> 5;
  if ((threadIdx.x & 31) == 0) { red[wave] = s; red[8 + wave] = s2; }
  __syncthreads();
  float S = 0.f, S2 = 0.f;
#pragma unroll
  for (int w = 0; w < 8; ++w) { S += red[w]; S2 += red[8 + w]; }
  float mean = S * (1.0f / D_MODEL);
  float var  = S2 * (1.0f / D_MODEL) - mean * mean;
  float rstd = rsqrtf(var + 1e-5f);
#pragma unroll
  for (int i = 0; i < 4; ++i) {
    int c = threadIdx.x + i * 256;
    float yv = (x[i] - mean) * rstd * gamma[c] + beta[c];
    if (outf) outf[base + c] = yv;
    if (outb) outb[base + c] = f2bf(yv);
  }
}

// ---------------------------------------------------------------- driver ----

extern "C" void kernel_launch(void* const* d_in, const int* in_sizes, int n_in,
                              void* d_out, int out_size, void* d_ws, size_t ws_size,
                              hipStream_t stream) {
  (void)in_sizes; (void)n_in; (void)out_size; (void)ws_size;
  const float* dec  = (const float*)d_in[0];
  const float* enc  = (const float*)d_in[1];
  const float* Wq_s = (const float*)d_in[2];  const float* bq_s = (const float*)d_in[3];
  const float* Wk_s = (const float*)d_in[4];  const float* bk_s = (const float*)d_in[5];
  const float* Wv_s = (const float*)d_in[6];  const float* bv_s = (const float*)d_in[7];
  const float* Wo_s = (const float*)d_in[8];  const float* bo_s = (const float*)d_in[9];
  const float* Wq_c = (const float*)d_in[10]; const float* bq_c = (const float*)d_in[11];
  const float* Wk_c = (const float*)d_in[12]; const float* bk_c = (const float*)d_in[13];
  const float* Wv_c = (const float*)d_in[14]; const float* bv_c = (const float*)d_in[15];
  const float* Wo_c = (const float*)d_in[16]; const float* bo_c = (const float*)d_in[17];
  const float* W1   = (const float*)d_in[18]; const float* b1   = (const float*)d_in[19];
  const float* W2   = (const float*)d_in[20]; const float* b2   = (const float*)d_in[21];
  const float* g1 = (const float*)d_in[22]; const float* be1 = (const float*)d_in[23];
  const float* g2 = (const float*)d_in[24]; const float* be2 = (const float*)d_in[25];
  const float* g3 = (const float*)d_in[26]; const float* be3 = (const float*)d_in[27];

  char* ws = (char*)d_ws;
  const size_t MB = 1ull << 20;
  unsigned short* wqsT = (unsigned short*)(ws + 0 * MB);
  unsigned short* wksT = (unsigned short*)(ws + 2 * MB);
  unsigned short* wvsT = (unsigned short*)(ws + 4 * MB);
  unsigned short* wosT = (unsigned short*)(ws + 6 * MB);
  unsigned short* wqcT = (unsigned short*)(ws + 8 * MB);
  unsigned short* wkcT = (unsigned short*)(ws + 10 * MB);
  unsigned short* wvcT = (unsigned short*)(ws + 12 * MB);
  unsigned short* wocT = (unsigned short*)(ws + 14 * MB);
  unsigned short* w1T  = (unsigned short*)(ws + 16 * MB);   // [4096][1024]
  unsigned short* w2T  = (unsigned short*)(ws + 24 * MB);   // [1024][4096]
  unsigned short* xb   = (unsigned short*)(ws + 32 * MB);   // current input bf16
  unsigned short* encb = (unsigned short*)(ws + 40 * MB);
  unsigned short* Qb   = (unsigned short*)(ws + 48 * MB);
  unsigned short* Kb   = (unsigned short*)(ws + 56 * MB);
  unsigned short* Vb   = (unsigned short*)(ws + 64 * MB);
  float*          Of   = (float*)(ws + 72 * MB);            // attn out f32
  unsigned short* Ob   = (unsigned short*)(ws + 88 * MB);
  float*          G    = (float*)(ws + 96 * MB);            // gemm out f32
  float*          x1   = (float*)(ws + 112 * MB);
  float*          x2   = (float*)(ws + 128 * MB);
  unsigned short* midb = (unsigned short*)(ws + 144 * MB);  // [4096][4096] bf16
  unsigned short* Vtg  = (unsigned short*)(ws + 176 * MB);  // [B*H][64][SEQ] bf16

  const dim3 blk256(256), blk128(128);
  const int nWD = (D_MODEL * D_MODEL) / 256;

  // ---- weight repack (bf16, B^T layouts)
  pack_heads<<<nWD, blk256, 0, stream>>>(Wq_s, wqsT);
  pack_heads<<<nWD, blk256, 0, stream>>>(Wk_s, wksT);
  pack_heads<<<nWD, blk256, 0, stream>>>(Wv_s, wvsT);
  pack_heads<<<nWD, blk256, 0, stream>>>(Wq_c, wqcT);
  pack_heads<<<nWD, blk256, 0, stream>>>(Wk_c, wkcT);
  pack_heads<<<nWD, blk256, 0, stream>>>(Wv_c, wvcT);
  transpose_cvt<<<nWD, blk256, 0, stream>>>(Wo_s, wosT, D_MODEL, D_MODEL);
  transpose_cvt<<<nWD, blk256, 0, stream>>>(Wo_c, wocT, D_MODEL, D_MODEL);
  transpose_cvt<<<(D_MODEL * D_FF) / 256, blk256, 0, stream>>>(W1, w1T, D_MODEL, D_FF);
  transpose_cvt<<<(D_FF * D_MODEL) / 256, blk256, 0, stream>>>(W2, w2T, D_FF, D_MODEL);

  const int nAct = NTOK * D_MODEL;
  cvt_bf16<<<nAct / 256, blk256, 0, stream>>>(dec, xb, nAct);
  cvt_bf16<<<nAct / 256, blk256, 0, stream>>>(enc, encb, nAct);

  const dim3 gProj(D_MODEL / BN, NTOK / BM);   // N=1024 GEMMs
  const dim3 gFF1(D_FF / BN, NTOK / BM);       // N=4096 GEMM
  const dim3 gAttn(SEQ / 64, BATCH * N_HEADS);

  // ---- masked self-attention
  gemm_bf16<0, false, true><<<gProj, blk256, 0, stream>>>(xb, wqsT, bq_s, nullptr, Qb, NTOK, D_MODEL, D_MODEL);
  gemm_bf16<0, false, true><<<gProj, blk256, 0, stream>>>(xb, wksT, bk_s, nullptr, Kb, NTOK, D_MODEL, D_MODEL);
  gemm_bf16<0, false, true><<<gProj, blk256, 0, stream>>>(xb, wvsT, bv_s, nullptr, Vb, NTOK, D_MODEL, D_MODEL);
  transpose_v_heads<<<nAct / 256, blk256, 0, stream>>>(Vb, Vtg);
  flash_attn<<<gAttn, blk128, 0, stream>>>(Qb, Kb, Vtg, Of, SEQ, 1);
  cvt_bf16<<<nAct / 256, blk256, 0, stream>>>(Of, Ob, nAct);
  gemm_bf16<0, true, false><<<gProj, blk256, 0, stream>>>(Ob, wosT, bo_s, G, nullptr, NTOK, D_MODEL, D_MODEL);
  resid_ln<<<NTOK, blk256, 0, stream>>>(dec, G, g1, be1, x1, xb);

  // ---- cross-attention
  gemm_bf16<0, false, true><<<gProj, blk256, 0, stream>>>(xb,   wqcT, bq_c, nullptr, Qb, NTOK, D_MODEL, D_MODEL);
  gemm_bf16<0, false, true><<<gProj, blk256, 0, stream>>>(encb, wkcT, bk_c, nullptr, Kb, NTOK, D_MODEL, D_MODEL);
  gemm_bf16<0, false, true><<<gProj, blk256, 0, stream>>>(encb, wvcT, bv_c, nullptr, Vb, NTOK, D_MODEL, D_MODEL);
  transpose_v_heads<<<nAct / 256, blk256, 0, stream>>>(Vb, Vtg);
  flash_attn<<<gAttn, blk128, 0, stream>>>(Qb, Kb, Vtg, Of, SEQ, 0);
  cvt_bf16<<<nAct / 256, blk256, 0, stream>>>(Of, Ob, nAct);
  gemm_bf16<0, true, false><<<gProj, blk256, 0, stream>>>(Ob, wocT, bo_c, G, nullptr, NTOK, D_MODEL, D_MODEL);
  resid_ln<<<NTOK, blk256, 0, stream>>>(x1, G, g2, be2, x2, xb);

  // ---- FFN
  gemm_bf16<1, false, true><<<gFF1, blk256, 0, stream>>>(xb, w1T, b1, nullptr, midb, NTOK, D_FF, D_MODEL);
  gemm_bf16<0, true, false><<<gProj, blk256, 0, stream>>>(midb, w2T, b2, G, nullptr, NTOK, D_MODEL, D_FF);
  resid_ln<<<NTOK, blk256, 0, stream>>>(x2, G, g3, be3, (float*)d_out, nullptr);
}
